// TemporalGAT_74045236183578
// MI455X (gfx1250) — matrix operand order repaired and checked by reading, not compiled, not run on hardware
//
#include <hip/hip_runtime.h>
#include <math.h>

#define HID 64
#define NHEAD 4
#define HD 16
#define NB 4
#define NW 12
#define NN 384
#define BN (NB*NN)
#define NPAIR (NB*NN*NN)   // 589824

typedef __attribute__((ext_vector_type(16))) _Float16 v16h;
typedef __attribute__((ext_vector_type(8)))  _Float16 v8h;
typedef __attribute__((ext_vector_type(8)))  float    v8f;

// ---------------------------------------------------------------- WMMA helpers
// A-matrix 16x32 f16 layout: lane m=L%16; lanes 0-15 hold K 0-15,
// lanes 16-31 hold K 16-31 (contiguous, matching B).
__device__ __forceinline__ v16h load_a_16x32(const _Float16* __restrict__ base,
                                             int ldk, int rowbase, int ktile, int lane) {
  int m  = lane & 15;
  int b0 = (lane < 16) ? 0 : 16;
  const _Float16* p = base + (rowbase + m) * ldk + ktile + b0;
  union { v16h v; v8h h[2]; } u;
  u.h[0] = *reinterpret_cast<const v8h*>(p);        // K = b0 .. b0+7
  u.h[1] = *reinterpret_cast<const v8h*>(p + 8);    // K = b0+8 .. b0+15
  return u.v;
}
// B-matrix 32x16 f16: B[k][o] = W[o][k] (W row-major [Nout][ldk]).
// col = lane%16; lanes 0-15 hold K 0-15, lanes 16-31 hold K 16-31 (contiguous).
__device__ __forceinline__ v16h load_b_32x16(const _Float16* __restrict__ w,
                                             int ldk, int coltile, int ktile, int lane) {
  int o  = coltile + (lane & 15);
  int kb = (lane < 16) ? 0 : 16;
  return *reinterpret_cast<const v16h*>(w + o * ldk + ktile + kb);
}
__device__ __forceinline__ v8f wmma_f16(v16h a, v16h b, v8f c) {
  return __builtin_amdgcn_wmma_f32_16x16x32_f16(false, a, false, b, (short)0, c, false, false);
}

// ---------------------------------------------------------------- LSTM
__global__ void lstm_kernel(const float* __restrict__ nf,
                            const float* __restrict__ Wih0, const float* __restrict__ Whh0,
                            const float* __restrict__ bih0, const float* __restrict__ bhh0,
                            const float* __restrict__ Wih1, const float* __restrict__ Whh1,
                            const float* __restrict__ bih1, const float* __restrict__ bhh1,
                            float* __restrict__ hout) {
  __shared__ float xs[8];
  __shared__ float hs[64];
  __shared__ float seq[NW * 64];
  int s = blockIdx.x;          // 0..BN-1 : sequence index (b*NN + n)
  int b = s / NN, n = s % NN;
  int u = threadIdx.x;         // 0..63 : hidden unit
  float c = 0.f, hcur = 0.f;
  hs[u] = 0.f;
  __syncthreads();
  // layer 0 (input dim 6)
  for (int t = 0; t < NW; ++t) {
    if (u < 6) xs[u] = nf[((b * NW + t) * NN + n) * 6 + u];
    __syncthreads();
    float g[4];
    #pragma unroll
    for (int q = 0; q < 4; ++q) {
      int r = q * 64 + u;
      float acc = bih0[r] + bhh0[r];
      #pragma unroll
      for (int k = 0; k < 6; ++k) acc += xs[k] * Wih0[r * 6 + k];
      const float* wr = &Whh0[r * 64];
      for (int k = 0; k < 64; ++k) acc += hs[k] * wr[k];
      g[q] = acc;
    }
    float ig = 1.f / (1.f + __expf(-g[0]));
    float fg = 1.f / (1.f + __expf(-g[1]));
    float gg = tanhf(g[2]);
    float og = 1.f / (1.f + __expf(-g[3]));
    c = fg * c + ig * gg;
    hcur = og * tanhf(c);
    __syncthreads();
    hs[u] = hcur; seq[t * 64 + u] = hcur;
    __syncthreads();
  }
  // layer 1 (input dim 64 = seq)
  c = 0.f; hcur = 0.f;
  hs[u] = 0.f;
  __syncthreads();
  for (int t = 0; t < NW; ++t) {
    const float* xrow = &seq[t * 64];
    float g[4];
    #pragma unroll
    for (int q = 0; q < 4; ++q) {
      int r = q * 64 + u;
      float acc = bih1[r] + bhh1[r];
      const float* wi = &Wih1[r * 64];
      for (int k = 0; k < 64; ++k) acc += xrow[k] * wi[k];
      const float* wh = &Whh1[r * 64];
      for (int k = 0; k < 64; ++k) acc += hs[k] * wh[k];
      g[q] = acc;
    }
    float ig = 1.f / (1.f + __expf(-g[0]));
    float fg = 1.f / (1.f + __expf(-g[1]));
    float gg = tanhf(g[2]);
    float og = 1.f / (1.f + __expf(-g[3]));
    c = fg * c + ig * gg;
    hcur = og * tanhf(c);
    __syncthreads();
    hs[u] = hcur;
    __syncthreads();
  }
  hout[s * 64 + u] = hcur;
}

// ---------------------------------------------------------------- f32 -> f16
__global__ void cvt_f32_f16(const float* __restrict__ s, _Float16* __restrict__ d, int n) {
  int i = blockIdx.x * blockDim.x + threadIdx.x;
  if (i < n) d[i] = (_Float16)s[i];
}

// ---------------------------------------------------------------- GAT: Wh = h @ W^T  (WMMA)
__global__ __launch_bounds__(256)
void gat_wh_kernel(const _Float16* __restrict__ h16, const _Float16* __restrict__ w16,
                   float* __restrict__ Wh) {
  int wave = (int)((blockIdx.x * blockDim.x + threadIdx.x) >> 5);
  int lane = threadIdx.x & 31;
  if (wave >= BN / 16) return;
  int rowbase = wave * 16;
  v8f acc[4] = {};
  #pragma unroll
  for (int kt = 0; kt < 64; kt += 32) {
    v16h a = load_a_16x32(h16, 64, rowbase, kt, lane);
    #pragma unroll
    for (int ct = 0; ct < 4; ++ct) {
      v16h bm = load_b_32x16(w16, 64, ct * 16, kt, lane);
      acc[ct] = wmma_f16(a, bm, acc[ct]);
    }
  }
  int m0 = (lane < 16) ? 0 : 8;
  int col = lane & 15;
  #pragma unroll
  for (int ct = 0; ct < 4; ++ct)
    #pragma unroll
    for (int r = 0; r < 8; ++r)
      Wh[(rowbase + m0 + r) * 64 + ct * 16 + col] = acc[ct][r];
}

// ---------------------------------------------------------------- GAT: e_src / e_dst
__global__ void gat_e_kernel(const float* __restrict__ Wh,
                             const float* __restrict__ asrc, const float* __restrict__ adst,
                             float* __restrict__ esrc, float* __restrict__ edst) {
  int idx = blockIdx.x * blockDim.x + threadIdx.x;   // (b, head, n)
  if (idx >= NB * NHEAD * NN) return;
  int n = idx % NN, hh = (idx / NN) % NHEAD, b = idx / (NN * NHEAD);
  const float* w = &Wh[(b * NN + n) * 64 + hh * 16];
  float s = 0.f, d2 = 0.f;
  #pragma unroll
  for (int k = 0; k < 16; ++k) { s += w[k] * asrc[hh * 16 + k]; d2 += w[k] * adst[hh * 16 + k]; }
  esrc[idx] = s; edst[idx] = d2;
}

// ---------------------------------------------------------------- GAT: softmax + alpha + h_gat
__global__ __launch_bounds__(256)
void gat_attn_kernel(const float* __restrict__ esrc, const float* __restrict__ edst,
                     const int* __restrict__ adj, const float* __restrict__ Wh,
                     float* __restrict__ alpha_out, float* __restrict__ hgat) {
  int wave = (int)((blockIdx.x * blockDim.x + threadIdx.x) >> 5);
  int lane = threadIdx.x & 31;
  if (wave >= NB * NHEAD * NN) return;
  int i = wave % NN, hh = (wave / NN) % NHEAD, b = wave / (NN * NHEAD);
  float es = esrc[wave];
  const float* ed = &edst[(b * NHEAD + hh) * NN];
  const int* am = &adj[(b * NN + i) * NN];
  // pass 1: row max
  float mx = -INFINITY;
  for (int j0 = 0; j0 < NN; j0 += 32) {
    int j = j0 + lane;
    float e = es + ed[j];
    e = e > 0.f ? e : 0.2f * e;
    if (am[j] == 0) e = -INFINITY;
    mx = fmaxf(mx, e);
  }
  #pragma unroll
  for (int o = 16; o > 0; o >>= 1) mx = fmaxf(mx, __shfl_xor(mx, o, 32));
  bool dead = (mx == -INFINITY);
  // pass 2: sum of exp
  float sum = 0.f;
  if (!dead) {
    for (int j0 = 0; j0 < NN; j0 += 32) {
      int j = j0 + lane;
      float e = es + ed[j];
      e = e > 0.f ? e : 0.2f * e;
      if (am[j] == 0) e = -INFINITY;
      sum += __expf(e - mx);
    }
    #pragma unroll
    for (int o = 16; o > 0; o >>= 1) sum += __shfl_xor(sum, o, 32);
  }
  float rinv = (sum > 0.f) ? 1.f / sum : 0.f;
  // pass 3: write alpha (nan_to_num -> 0 for fully-masked rows)
  float* arow = &alpha_out[((size_t)(b * NHEAD + hh) * NN + i) * NN];
  for (int j0 = 0; j0 < NN; j0 += 32) {
    int j = j0 + lane;
    float a_ = 0.f;
    if (!dead) {
      float e = es + ed[j];
      e = e > 0.f ? e : 0.2f * e;
      if (am[j] == 0) e = -INFINITY;
      a_ = __expf(e - mx) * rinv;
    }
    arow[j] = a_;
  }
  // phase B: h_gat[b,i,hh,:] = sum_j alpha_j * Wh[b,j,hh,:]
  int d = lane & 15, half = lane >> 4;
  float acc = 0.f;
  for (int j = half; j < NN; j += 2)
    acc += arow[j] * Wh[(b * NN + j) * 64 + hh * 16 + d];
  acc += __shfl_xor(acc, 16, 32);
  if (lane < 16) hgat[(b * NN + i) * 64 + hh * 16 + d] = acc;
}

// ---------------------------------------------------------------- residual + LN + ELU (in place)
__global__ void ln_kernel(float* __restrict__ h, const float* __restrict__ hgat,
                          const float* __restrict__ g, const float* __restrict__ bb) {
  __shared__ float red[64];
  int node = blockIdx.x, u = threadIdx.x;
  float v = h[node * 64 + u] + hgat[node * 64 + u];
  red[u] = v; __syncthreads();
  for (int o = 32; o > 0; o >>= 1) { if (u < o) red[u] += red[u + o]; __syncthreads(); }
  float mean = red[0] * (1.f / 64.f);
  __syncthreads();
  float dv = v - mean;
  red[u] = dv * dv; __syncthreads();
  for (int o = 32; o > 0; o >>= 1) { if (u < o) red[u] += red[u + o]; __syncthreads(); }
  float var = red[0] * (1.f / 64.f);
  float y = dv * rsqrtf(var + 1e-5f) * g[u] + bb[u];
  h[node * 64 + u] = y > 0.f ? y : (__expf(y) - 1.f);
}

// ---------------------------------------------------------------- edge MLP (dominant WMMA kernel)
__global__ __launch_bounds__(256)
void edge_mlp_kernel(const _Float16* __restrict__ h16, const float* __restrict__ ef,
                     const float* __restrict__ epW, const float* __restrict__ epb,
                     const _Float16* __restrict__ W1h, const float* __restrict__ b1,
                     const _Float16* __restrict__ W2h, const float* __restrict__ b2,
                     const _Float16* __restrict__ W3h, const float* __restrict__ b3,
                     float* __restrict__ out) {
  __shared__ __align__(32) _Float16 lds[8 * 3072];   // 48 KB: 16x192 f16 per wave
  int wv = threadIdx.x >> 5;
  int lane = threadIdx.x & 31;
  _Float16* A = &lds[wv * 3072];
  int P0 = (blockIdx.x * 8 + wv) * 16;               // grid covers NPAIR exactly
  int r = lane & 15, half = lane >> 4;
  int pair = P0 + r;
  int b = pair / (NN * NN);
  int rem = pair % (NN * NN);
  int i = rem / NN, j = rem % NN;

  // build A row r: [h_i(64) | h_j(64) | edge_proj(edge_last[b,i,j,:5])(64)]
  {
    // vectorized f16 copy of node embeddings: 8x b128 load + 8x b128 ds store
    int node = (half == 0) ? (b * NN + i) : (b * NN + j);
    const v8h* hsrc = reinterpret_cast<const v8h*>(&h16[node * 64]);
    v8h* hdst = reinterpret_cast<v8h*>(&A[r * 192 + half * 64]);
    #pragma unroll
    for (int c = 0; c < 8; ++c) hdst[c] = hsrc[c];

    // edge projection: 32 outputs per lane, packed into v8h (4x b128 ds store)
    const float* e5 = &ef[(size_t)(((b * NW + (NW - 1)) * NN + i) * NN + j) * 5];
    float e0 = e5[0], e1 = e5[1], e2 = e5[2], e3 = e5[3], e4 = e5[4];
    int obase = half * 32;
    #pragma unroll
    for (int og = 0; og < 4; ++og) {
      v8h pack;
      #pragma unroll
      for (int t = 0; t < 8; ++t) {
        int o = obase + og * 8 + t;
        const float* w = &epW[o * 5];
        float v = e0 * w[0] + e1 * w[1] + e2 * w[2] + e3 * w[3] + e4 * w[4] + epb[o];
        pack[t] = (_Float16)v;
      }
      *reinterpret_cast<v8h*>(&A[r * 192 + 128 + obase + og * 8]) = pack;
    }
  }
  // layer 1: 192 -> 64
  v8f acc[4] = {};
  #pragma unroll
  for (int kt = 0; kt < 192; kt += 32) {
    v16h a = load_a_16x32(A, 192, 0, kt, lane);
    #pragma unroll
    for (int ct = 0; ct < 4; ++ct) {
      v16h bm = load_b_32x16(W1h, 192, ct * 16, kt, lane);
      acc[ct] = wmma_f16(a, bm, acc[ct]);
    }
  }
  int m0 = half * 8, col = r;
  _Float16* A2 = A;                 // reuse (LDS ops are in-order within a wave)
  #pragma unroll
  for (int ct = 0; ct < 4; ++ct) {
    float bias = b1[ct * 16 + col];
    #pragma unroll
    for (int rr = 0; rr < 8; ++rr) {
      float v = acc[ct][rr] + bias;
      A2[(rr + m0) * 64 + ct * 16 + col] = (_Float16)(v > 0.f ? v : 0.f);
    }
  }
  // layer 2: 64 -> 32
  v8f acc2[2] = {};
  #pragma unroll
  for (int kt = 0; kt < 64; kt += 32) {
    v16h a = load_a_16x32(A2, 64, 0, kt, lane);
    #pragma unroll
    for (int ct = 0; ct < 2; ++ct) {
      v16h bm = load_b_32x16(W2h, 64, ct * 16, kt, lane);
      acc2[ct] = wmma_f16(a, bm, acc2[ct]);
    }
  }
  _Float16* A3 = A + 1024;          // 16x32, disjoint from A2 region
  #pragma unroll
  for (int ct = 0; ct < 2; ++ct) {
    float bias = b2[ct * 16 + col];
    #pragma unroll
    for (int rr = 0; rr < 8; ++rr) {
      float v = acc2[ct][rr] + bias;
      A3[(rr + m0) * 32 + ct * 16 + col] = (_Float16)(v > 0.f ? v : 0.f);
    }
  }
  // layer 3: 32 -> 1, sigmoid
  if (half == 0) {
    float s = b3[0];
    for (int k = 0; k < 32; ++k) s += (float)A3[r * 32 + k] * (float)W3h[k];
    out[pair] = 1.f / (1.f + __expf(-s));
  }
}

// ---------------------------------------------------------------- launcher
extern "C" void kernel_launch(void* const* d_in, const int* in_sizes, int n_in,
                              void* d_out, int out_size, void* d_ws, size_t ws_size,
                              hipStream_t stream) {
  (void)in_sizes; (void)n_in; (void)out_size; (void)ws_size;
  const float* nf   = (const float*)d_in[0];
  const float* ef   = (const float*)d_in[1];
  const int*   adj  = (const int*)d_in[2];
  const float* Wih0 = (const float*)d_in[3];
  const float* Whh0 = (const float*)d_in[4];
  const float* bih0 = (const float*)d_in[5];
  const float* bhh0 = (const float*)d_in[6];
  const float* Wih1 = (const float*)d_in[7];
  const float* Whh1 = (const float*)d_in[8];
  const float* bih1 = (const float*)d_in[9];
  const float* bhh1 = (const float*)d_in[10];
  const float* gat0W = (const float*)d_in[11];
  const float* gat0as = (const float*)d_in[12];
  const float* gat0ad = (const float*)d_in[13];
  const float* gat1W = (const float*)d_in[14];
  const float* gat1as = (const float*)d_in[15];
  const float* gat1ad = (const float*)d_in[16];
  const float* ln0g = (const float*)d_in[17];
  const float* ln0b = (const float*)d_in[18];
  const float* ln1g = (const float*)d_in[19];
  const float* ln1b = (const float*)d_in[20];
  const float* epW = (const float*)d_in[21];
  const float* epb = (const float*)d_in[22];
  const float* W1 = (const float*)d_in[23];
  const float* b1 = (const float*)d_in[24];
  const float* W2 = (const float*)d_in[25];
  const float* b2 = (const float*)d_in[26];
  const float* W3 = (const float*)d_in[27];
  const float* b3 = (const float*)d_in[28];

  float* out_logits = (float*)d_out;
  float* out_alpha  = out_logits + (size_t)NPAIR;

  // workspace bump allocator (64B aligned)
  char* wsp = (char*)d_ws;
  auto alloc = [&](size_t bytes) { void* p = (void*)wsp; wsp += (bytes + 63) & ~(size_t)63; return p; };
  float*     hbuf = (float*)alloc((size_t)BN * 64 * 4);
  float*     hgat = (float*)alloc((size_t)BN * 64 * 4);
  float*     Whb  = (float*)alloc((size_t)BN * 64 * 4);
  float*     esrc = (float*)alloc((size_t)NB * NHEAD * NN * 4);
  float*     edst = (float*)alloc((size_t)NB * NHEAD * NN * 4);
  _Float16*  h16  = (_Float16*)alloc((size_t)BN * 64 * 2);
  _Float16*  gw16 = (_Float16*)alloc((size_t)64 * 64 * 2);
  _Float16*  W1h  = (_Float16*)alloc((size_t)64 * 192 * 2);
  _Float16*  W2h  = (_Float16*)alloc((size_t)32 * 64 * 2);
  _Float16*  W3h  = (_Float16*)alloc((size_t)64 * 2);

  // 1) LSTM encoder -> h (BN x 64)
  lstm_kernel<<<BN, 64, 0, stream>>>(nf, Wih0, Whh0, bih0, bhh0, Wih1, Whh1, bih1, bhh1, hbuf);

  // 2) weight precision prep (f32 -> f16) for WMMA
  cvt_f32_f16<<<(64 * 192 + 255) / 256, 256, 0, stream>>>(W1, W1h, 64 * 192);
  cvt_f32_f16<<<(32 * 64 + 255) / 256, 256, 0, stream>>>(W2, W2h, 32 * 64);
  cvt_f32_f16<<<1, 32, 0, stream>>>(W3, W3h, 32);

  // 3) two GAT layers
  const float* gatW[2]  = { gat0W, gat1W };
  const float* gatAS[2] = { gat0as, gat1as };
  const float* gatAD[2] = { gat0ad, gat1ad };
  const float* lng[2]   = { ln0g, ln1g };
  const float* lnb[2]   = { ln0b, ln1b };
  for (int l = 0; l < 2; ++l) {
    cvt_f32_f16<<<(BN * 64 + 255) / 256, 256, 0, stream>>>(hbuf, h16, BN * 64);
    cvt_f32_f16<<<(64 * 64 + 255) / 256, 256, 0, stream>>>(gatW[l], gw16, 64 * 64);
    gat_wh_kernel<<<(BN / 16 + 7) / 8, 256, 0, stream>>>(h16, gw16, Whb);
    gat_e_kernel<<<(NB * NHEAD * NN + 255) / 256, 256, 0, stream>>>(Whb, gatAS[l], gatAD[l], esrc, edst);
    gat_attn_kernel<<<(NB * NHEAD * NN + 7) / 8, 256, 0, stream>>>(esrc, edst, adj, Whb, out_alpha, hgat);
    ln_kernel<<<BN, 64, 0, stream>>>(hbuf, hgat, lng[l], lnb[l]);
  }

  // final h -> f16 once, for vectorized A-tile staging in the edge MLP
  cvt_f32_f16<<<(BN * 64 + 255) / 256, 256, 0, stream>>>(hbuf, h16, BN * 64);

  // 4) edge MLP (WMMA): 589824 pairs, 16 per wave, 8 waves per block
  edge_mlp_kernel<<<NPAIR / (16 * 8), 256, 0, stream>>>(h16, ef, epW, epb,
                                                        W1h, b1, W2h, b2, W3h, b3,
                                                        out_logits);
}
